// FFM_77936476553484
// MI455X (gfx1250) — matrix operand order repaired and checked by reading, not compiled, not run on hardware
//
#include <hip/hip_runtime.h>
#include <math.h>

// ---------------------------------------------------------------------------
// FFM inference:  out = sigmoid( x@w_lin + b + 0.5 * rowsum((x@W) * x) )
// W (64x64) is batch-independent -> built once (pre-swizzled bf16 hi/lo
// fragments, 0.5 folded in), then the main kernel is a streaming
// (B x 64) @ (64 x 64) bf16-split WMMA GEMM fused with row reduction +
// sigmoid.  w_lin is folded into the WMMA accumulator init.
// x tiles travel global -> LDS via GLOBAL_LOAD_ASYNC_TO_LDS_B128 (inline asm,
// ASYNCcnt-tracked, no VGPR round trip).  Memory-bound ~12us @ 23.3 TB/s.
// ---------------------------------------------------------------------------

typedef __attribute__((ext_vector_type(16))) __bf16          v16bf;
typedef __attribute__((ext_vector_type(16))) unsigned short  v16us;
typedef __attribute__((ext_vector_type(8)))  unsigned int    v8u;
typedef __attribute__((ext_vector_type(8)))  float           v8f;

#define NFEAT   64
#define NFIELD  8
#define NH      16
#define TILE_M  16
#define WAVES_PER_BLOCK 8
#define NFRAG   256            // 4 nb * 2 kc * 32 lanes fragment entries

__device__ __forceinline__ float trunc16f(float f) {
    unsigned u = __builtin_bit_cast(unsigned, f) & 0xFFFF0000u;
    return __builtin_bit_cast(float, u);
}

__device__ __forceinline__ v8f wmma_bf16(v16bf a, v16bf b, v8f c) {
    // D = A(16x32 bf16) * B(32x16 bf16) + C(16x16 f32)
    return __builtin_amdgcn_wmma_f32_16x16x32_bf16(
        /*neg_a=*/false, a, /*neg_b=*/false, b,
        /*c_mod=*/(short)0, c, /*reuse_a=*/false, /*reuse_b=*/false);
}

// ---------------------------------------------------------------------------
// Kernel 1: W'[k][n] = 0.5 * (f_k != f_n) * sum_h v[k][f_n][h]*v[n][f_k][h]
// then emit pre-swizzled bf16 hi/lo B-operand fragments into ws:
//   fragHi[((nb*2+kc)*32 + lane)*16 + e]  (ushort), fragLo at +4096
//   element e <-> K = kc*32 + (lane>>4)*16 + e, column N = nb*16 + (lane&15)
// ---------------------------------------------------------------------------
__global__ void ffm_build_w(const float* __restrict__ v,
                            const int*   __restrict__ fields,
                            unsigned short* __restrict__ frag) {
    __shared__ int   fld[NFEAT];
    __shared__ float Wsh[NFEAT * NFEAT];
    if (threadIdx.x < NFEAT) fld[threadIdx.x] = fields[threadIdx.x];
    __syncthreads();
    for (int p = threadIdx.x; p < NFEAT * NFEAT; p += blockDim.x) {
        int i = p >> 6, j = p & 63;
        int gi = fld[i], gj = fld[j];
        float s = 0.0f;
        if (gi != gj) {
            const float* vi = v + i * (NFIELD * NH) + gj * NH;
            const float* vj = v + j * (NFIELD * NH) + gi * NH;
#pragma unroll
            for (int h = 0; h < NH; ++h) s = fmaf(vi[h], vj[h], s);
        }
        Wsh[p] = 0.5f * s;
    }
    __syncthreads();
    // one thread per (nb, kc, lane) fragment slice
    {
        const int tid  = threadIdx.x;           // 0..255
        const int lane = tid & 31;
        const int kc   = (tid >> 5) & 1;
        const int nb   = tid >> 6;
        const int half = lane >> 4;
        const int nl   = lane & 15;
        unsigned short* fragHi = frag;
        unsigned short* fragLo = frag + 16 * NFRAG;
        const int fi = ((nb * 2 + kc) * 32 + lane) * 16;
#pragma unroll
        for (int e = 0; e < 16; ++e) {
            float f = Wsh[(kc * 32 + half * 16 + e) * NFEAT + nb * 16 + nl];
            unsigned u = __builtin_bit_cast(unsigned, f);
            fragHi[fi + e] = (unsigned short)(u >> 16);
            float l = f - trunc16f(f);
            fragLo[fi + e] = (unsigned short)(__builtin_bit_cast(unsigned, l) >> 16);
        }
    }
}

// ---------------------------------------------------------------------------
// Kernel 2: streaming fused GEMM + reduction + sigmoid
// One wave handles one 16-row tile per iteration (grid-stride over tiles).
// ---------------------------------------------------------------------------
__global__ void __launch_bounds__(256)
ffm_main(const float* __restrict__ x,
         const float* __restrict__ w_lin,
         const float* __restrict__ b_lin,
         const unsigned short* __restrict__ Wfrag,
         float*       __restrict__ out,
         int n_tiles) {
    __shared__ __align__(16) float lds[WAVES_PER_BLOCK][TILE_M * NFEAT];

    const int tid  = threadIdx.x;
    const int lane = tid & 31;
    const int warp = tid >> 5;
    const int half = lane >> 4;   // which half-wave
    const int nl   = lane & 15;
    float* tileLds = &lds[warp][0];

    // ---- loop-invariant: pre-swizzled W fragments, vectorized 32B loads
    const v16us* fragH = (const v16us*)Wfrag;         // 256 entries
    const v16us* fragL = fragH + NFRAG;
    v16bf bhi[4][2], blo[4][2];
#pragma unroll
    for (int nb = 0; nb < 4; ++nb) {
#pragma unroll
        for (int kc = 0; kc < 2; ++kc) {
            const int idx = (nb * 2 + kc) * 32 + lane;
            bhi[nb][kc] = __builtin_bit_cast(v16bf, fragH[idx]);
            blo[nb][kc] = __builtin_bit_cast(v16bf, fragL[idx]);
        }
    }
    float wl[4];
#pragma unroll
    for (int nb = 0; nb < 4; ++nb) wl[nb] = w_lin[nb * 16 + nl];
    const float bias = b_lin[0];

    const int gwave  = blockIdx.x * WAVES_PER_BLOCK + warp;
    const int stride = gridDim.x * WAVES_PER_BLOCK;

    for (int t = gwave; t < n_tiles; t += stride) {
        // ---- 16x64 f32 tile (4KB) -> this wave's LDS slab via async DMA.
        // VDST = wave-relative LDS byte offset (low 32 bits of generic LDS
        // pointer), VADDR = 64-bit global address, tracked by ASYNCcnt.
        {
            const unsigned ldsOff = (unsigned)(uintptr_t)tileLds;
            const unsigned long long gbase =
                (unsigned long long)(uintptr_t)((const char*)x + (size_t)t * 4096);
#pragma unroll
            for (int i = 0; i < 8; ++i) {
                const unsigned off = (unsigned)(i * 512 + lane * 16);
                asm volatile("global_load_async_to_lds_b128 %0, %1, off"
                             :
                             : "v"(ldsOff + off),
                               "v"(gbase + (unsigned long long)off)
                             : "memory");
            }
        }
        // prefetch next tile while we compute (global_prefetch_b8)
        if (t + stride < n_tiles) {
            const char* pre = (const char*)x + (size_t)(t + stride) * 4096 + (size_t)lane * 128;
            __builtin_prefetch(pre, 0, 1);
        }
        // drain the async tile copy before LDS reads
        asm volatile("s_wait_asynccnt 0" ::: "memory");

        // ---- A fragments (16x32 bf16 layout) via truncation split:
        //   hi = top 16 bits of f32 (exact),  lo = bf16(f - trunc16(f))
        //   hi pack: v_perm_b32 (1/2 elems); lo: f - trunc16(f), then v_perm_b32 (1/2 elems)
        // element e <-> K = kc*32 + half*16 + e, row M = nl
        v16bf ahi[2], alo[2];
#pragma unroll
        for (int kc = 0; kc < 2; ++kc) {
            const float* base = tileLds + nl * NFEAT + kc * 32 + half * 16;
            float f[16];
            float4 q0 = *(const float4*)(base);
            float4 q1 = *(const float4*)(base + 4);
            float4 q2 = *(const float4*)(base + 8);
            float4 q3 = *(const float4*)(base + 12);
            f[0]=q0.x; f[1]=q0.y; f[2]=q0.z; f[3]=q0.w;
            f[4]=q1.x; f[5]=q1.y; f[6]=q1.z; f[7]=q1.w;
            f[8]=q2.x; f[9]=q2.y; f[10]=q2.z; f[11]=q2.w;
            f[12]=q3.x; f[13]=q3.y; f[14]=q3.z; f[15]=q3.w;
            v8u hi, lo;
#pragma unroll
            for (int p = 0; p < 8; ++p) {
                float a0 = f[2 * p], a1 = f[2 * p + 1];
                hi[p] = __builtin_amdgcn_perm(
                    __builtin_bit_cast(unsigned, a1),
                    __builtin_bit_cast(unsigned, a0), 0x07060302u);
                float l0 = a0 - trunc16f(a0);
                float l1 = a1 - trunc16f(a1);
                lo[p] = __builtin_amdgcn_perm(
                    __builtin_bit_cast(unsigned, l1),
                    __builtin_bit_cast(unsigned, l0), 0x07060302u);
            }
            ahi[kc] = __builtin_bit_cast(v16bf, hi);
            alo[kc] = __builtin_bit_cast(v16bf, lo);
        }

        // ---- P = wl + X_tile @ (0.5*W) via 3-product bf16 split:
        //      X@W ~= Xhi@Whi + Xlo@Whi + Xhi@Wlo       (24 WMMAs / tile)
        //      accumulator initialized with w_lin (broadcast per column)
        v8f acc[4];
#pragma unroll
        for (int nb = 0; nb < 4; ++nb) {
            v8f c = wl[nb];          // splat: C[m][n] = w_lin[n]
#pragma unroll
            for (int kc = 0; kc < 2; ++kc) {
                c = wmma_bf16(ahi[kc], bhi[nb][kc], c);
                c = wmma_bf16(alo[kc], bhi[nb][kc], c);
                c = wmma_bf16(ahi[kc], blo[nb][kc], c);
            }
            acc[nb] = c;
        }

        // ---- fused row reduction: z_m = sum_n x[m][n]*acc[m][n] + b
        // D layout: VGPR r <-> row (r + 8*half), lane&15 <-> col within 16-tile
        float* outp = out + (size_t)t * TILE_M;
#pragma unroll
        for (int r = 0; r < 8; ++r) {
            const int mr = half * 8 + r;
            float s = 0.0f;
#pragma unroll
            for (int nb = 0; nb < 4; ++nb) {
                float xv = tileLds[mr * NFEAT + nb * 16 + nl];
                s = fmaf(xv, acc[nb][r], s);
            }
            // reduce across the 16 lanes of this half-wave
            s += __shfl_xor(s, 1);
            s += __shfl_xor(s, 2);
            s += __shfl_xor(s, 4);
            s += __shfl_xor(s, 8);
            if (nl == 0) {
                float z = s + bias;
                outp[mr] = 1.0f / (1.0f + __expf(-z));
            }
        }
    }
}

// ---------------------------------------------------------------------------
extern "C" void kernel_launch(void* const* d_in, const int* in_sizes, int n_in,
                              void* d_out, int out_size, void* d_ws, size_t ws_size,
                              hipStream_t stream) {
    const float* x     = (const float*)d_in[0];   // (B, 64)
    const float* w_lin = (const float*)d_in[1];   // (64,)
    const float* b_lin = (const float*)d_in[2];   // (1,)
    const float* v     = (const float*)d_in[3];   // (64, 8, 16)
    const int*   flds  = (const int*)  d_in[4];   // (64,)
    float*       outp  = (float*)d_out;

    const int B       = in_sizes[0] / NFEAT;
    const int n_tiles = B / TILE_M;

    unsigned short* Wfrag = (unsigned short*)d_ws;  // 16 KB scratch

    ffm_build_w<<<1, 256, 0, stream>>>(v, flds, Wfrag);

    int blocks = 1024;
    int max_blocks = (n_tiles + WAVES_PER_BLOCK - 1) / WAVES_PER_BLOCK;
    if (blocks > max_blocks) blocks = max_blocks;
    if (blocks < 1) blocks = 1;
    ffm_main<<<blocks, 256, 0, stream>>>(x, w_lin, b_lin, Wfrag, outp, n_tiles);
}
